// NeuroBackMamba_13134009991208
// MI455X (gfx1250) — matrix operand order repaired and checked by reading, not compiled, not run on hardware
//
#include <hip/hip_runtime.h>
#include <hip/hip_bf16.h>
#include <math.h>

// ---------------------------------------------------------------- constants
#define B_    8
#define N_    2048
#define IN_   128
#define H_    256
#define DS_   64
#define DC_   4
#define DI_   512        // EXP * H
#define DTR_  16
#define E_    262144
#define M_    (B_ * N_)  // 16384 rows

typedef __bf16 bf16t;
typedef __attribute__((ext_vector_type(16))) __bf16 v16bf;
typedef __attribute__((ext_vector_type(8)))  float  v8f;

__device__ __forceinline__ float silu_f(float x) {
    return x * (1.0f / (1.0f + __expf(-x)));
}

// Async global->LDS copy (gfx1250, ASYNCcnt-tracked). LDS operand is the
// wave-relative LDS byte address = low 32 bits of the generic pointer.
__device__ __forceinline__ void async_ld_b128(void* lds_dst, const void* gsrc) {
    unsigned lds = (unsigned)(unsigned long long)(uintptr_t)lds_dst;
    unsigned long long ga = (unsigned long long)(uintptr_t)gsrc;
    asm volatile("global_load_async_to_lds_b128 %0, %1, off"
                 :: "v"(lds), "v"(ga) : "memory");
}
__device__ __forceinline__ void wait_asynccnt0() {
    asm volatile("s_wait_asynccnt 0x0" ::: "memory");
}

// ================================================================= GEMM
// C[M,Nn](f32, ldc, +colOff) = A[M,K](f32) @ W[K,Nn](f32) (+bias)(+gelu)
// bf16 WMMA with on-the-fly f32->bf16 conversion into LDS (weights/acts
// live in the 192MB L2, so re-reading f32 tiles is cheap).
// Block = 256 threads = 8 wave32; tile M=128, N=64, K-step 32.
__global__ __launch_bounds__(256) void gemm_bf16_wmma(
    const float* __restrict__ A, int lda,
    const float* __restrict__ W, int ldw,
    const float* __restrict__ bias,
    float* __restrict__ C, int ldc, int colOff,
    int K, int Nn, int epi)
{
    __shared__ bf16t sA[128 * 34];   // [row][k], pad 2
    __shared__ bf16t sW[64 * 34];    // [n][k] (transposed), pad 2

    const int tid  = threadIdx.x;
    const int wave = tid >> 5;
    const int lane = tid & 31;
    const int lh   = lane >> 4;      // lane half
    const int lr   = lane & 15;
    const int mBase = blockIdx.x * 128;
    const int nBase = blockIdx.y * 64;

    v8f acc[4];
#pragma unroll
    for (int g = 0; g < 4; ++g)
#pragma unroll
        for (int i = 0; i < 8; ++i) acc[g][i] = 0.0f;

    for (int k0 = 0; k0 < K; k0 += 32) {
        __syncthreads();
        // Stage A tile: 128x32 f32 -> bf16 (1024 float4 loads)
#pragma unroll
        for (int i = 0; i < 4; ++i) {
            int idx = tid + i * 256;
            int row = idx >> 3;
            int c4  = idx & 7;
            const float* src = A + (size_t)(mBase + row) * lda + k0 + c4 * 4;
            float4 v = *(const float4*)src;
            if (k0 + 32 < K) __builtin_prefetch(src + 32, 0, 1);
            bf16t* dst = &sA[row * 34 + c4 * 4];
            dst[0] = (bf16t)v.x; dst[1] = (bf16t)v.y;
            dst[2] = (bf16t)v.z; dst[3] = (bf16t)v.w;
        }
        // Stage W tile transposed: sW[n][k]; mask n >= Nn (x_proj N=144)
#pragma unroll
        for (int i = 0; i < 8; ++i) {
            int idx = tid + i * 256;
            int kk  = idx >> 6;
            int nn  = idx & 63;
            int n   = nBase + nn;
            float val = (n < Nn) ? W[(size_t)(k0 + kk) * ldw + n] : 0.0f;
            sW[nn * 34 + kk] = (bf16t)val;
        }
        __syncthreads();

        // A fragment: ISA 16-bit A 16x32 layout.
        // vv<4 -> K = 2vv + 8*lh (0..7 / 8..15); vv>=4 -> K = 8+2vv + 8*lh (16..23 / 24..31)
        union { v16bf v; unsigned u[8]; } fa;
        const bf16t* arow = &sA[(wave * 16 + lr) * 34];
#pragma unroll
        for (int vv = 0; vv < 8; ++vv) {
            int k = (vv < 4 ? 2 * vv : 8 + 2 * vv) + 8 * lh;
            fa.u[vv] = *(const unsigned*)(arow + k);
        }
        // 4 WMMAs covering N=64 (col = g*16 + lr; K-major per column)
#pragma unroll
        for (int g = 0; g < 4; ++g) {
            union { v16bf v; unsigned u[8]; } fb;
            const bf16t* bcol = &sW[(g * 16 + lr) * 34];
#pragma unroll
            for (int vv = 0; vv < 8; ++vv)
                fb.u[vv] = *(const unsigned*)(bcol + (vv < 4 ? 2 * vv : 8 + 2 * vv) + 8 * lh);
            acc[g] = __builtin_amdgcn_wmma_f32_16x16x32_bf16(
                false, fa.v, false, fb.v, (short)0, acc[g], false, false);
        }
    }

    // Epilogue: C layout -> VGPR j: M = j + 8*lh, N = lane&15
#pragma unroll
    for (int g = 0; g < 4; ++g) {
#pragma unroll
        for (int j = 0; j < 8; ++j) {
            int row = mBase + wave * 16 + j + 8 * lh;
            int col = nBase + g * 16 + lr;
            if (col < Nn) {
                float v = acc[g][j];
                if (bias) v += bias[col];
                if (epi == 1) {  // tanh-approx GELU (jax.nn.gelu default)
                    float x = v;
                    float c = 0.7978845608028654f * (x + 0.044715f * x * x * x);
                    v = 0.5f * x * (1.0f + tanhf(c));
                }
                C[(size_t)row * ldc + colOff + col] = v;
            }
        }
    }
}

// ============================================================ pointwise ops
__global__ void zero_kernel(float* p, int n) {
    int i = blockIdx.x * 256 + threadIdx.x;
    if (i < n) p[i] = 0.0f;
}

__global__ void degree_kernel(const int* __restrict__ eidx, float* __restrict__ deg) {
    int e = blockIdx.x * 256 + threadIdx.x;
    if (e < E_) atomicAdd(&deg[eidx[e]], 1.0f);
}

// Stable descending argsort of integer degrees (per batch) via bitonic sort
// on 64-bit keys ((~deg)<<32 | idx) in LDS.
__global__ __launch_bounds__(1024) void sort_kernel(const float* __restrict__ deg,
                                                    int* __restrict__ sortidx) {
    __shared__ unsigned long long key[N_];
    const int b = blockIdx.x;
    const int tid = threadIdx.x;
    for (int i = tid; i < N_; i += 1024) {
        unsigned c = (unsigned)deg[b * N_ + i];
        key[i] = ((unsigned long long)(0xFFFFFFFFu - c) << 32) | (unsigned)i;
    }
    __syncthreads();
    for (int k = 2; k <= N_; k <<= 1) {
        for (int j = k >> 1; j > 0; j >>= 1) {
            for (int i = tid; i < N_; i += 1024) {
                int ixj = i ^ j;
                if (ixj > i) {
                    unsigned long long a = key[i], c = key[ixj];
                    bool asc = ((i & k) == 0);
                    if (asc ? (a > c) : (a < c)) { key[i] = c; key[ixj] = a; }
                }
            }
            __syncthreads();
        }
    }
    for (int i = tid; i < N_; i += 1024)
        sortidx[b * N_ + i] = (int)(key[i] & 0xFFFFFFFFu);
}

__global__ void invert_kernel(const int* __restrict__ sortidx, int* __restrict__ undo) {
    int gid = blockIdx.x * 256 + threadIdx.x;
    if (gid < M_) {
        int b = gid >> 11;
        undo[b * N_ + sortidx[gid]] = gid & (N_ - 1);
    }
}

__global__ void gather_kernel(const float* __restrict__ in, const int* __restrict__ idx,
                              float* __restrict__ out) {
    int gid = blockIdx.x * 256 + threadIdx.x;   // M_*H_ elements
    int c = gid & (H_ - 1);
    int r = gid >> 8;
    int b = r >> 11;
    int src = b * N_ + idx[r];
    out[gid] = in[(size_t)src * H_ + c];
}

// LayerNorm over H=256, one wave32 per row; optional residual input R.
__global__ __launch_bounds__(256) void ln_kernel(const float* __restrict__ A,
                                                 const float* __restrict__ R,
                                                 const float* __restrict__ g,
                                                 const float* __restrict__ bta,
                                                 float* __restrict__ out) {
    int lane = threadIdx.x & 31;
    int row = blockIdx.x * 8 + (threadIdx.x >> 5);
    float v[8];
    float sum = 0.0f;
#pragma unroll
    for (int i = 0; i < 8; ++i) {
        int c = i * 32 + lane;
        float x = A[(size_t)row * H_ + c];
        if (R) x += R[(size_t)row * H_ + c];
        v[i] = x;
        sum += x;
    }
#pragma unroll
    for (int off = 16; off > 0; off >>= 1) sum += __shfl_xor(sum, off, 32);
    float mean = sum * (1.0f / H_);
    float sq = 0.0f;
#pragma unroll
    for (int i = 0; i < 8; ++i) { float d = v[i] - mean; sq += d * d; }
#pragma unroll
    for (int off = 16; off > 0; off >>= 1) sq += __shfl_xor(sq, off, 32);
    float inv = rsqrtf(sq * (1.0f / H_) + 1e-5f);
#pragma unroll
    for (int i = 0; i < 8; ++i) {
        int c = i * 32 + lane;
        out[(size_t)row * H_ + c] = (v[i] - mean) * inv * g[c] + bta[c];
    }
}

// Depthwise causal conv (DC=4) + bias + SiLU. rev flag folds the sequence
// flip of the backward branch into the tap offsets (no materialized flip).
__global__ void conv_silu_kernel(const float* __restrict__ xz,
                                 const float* __restrict__ w,
                                 const float* __restrict__ bias,
                                 float* __restrict__ xi, int rev) {
    int gid = blockIdx.x * 256 + threadIdx.x;   // M_*DI_
    int d = gid & (DI_ - 1);
    int r = gid >> 9;
    int b = r >> 11;
    int t = r & (N_ - 1);
    float acc = bias[d];
#pragma unroll
    for (int k = 0; k < DC_; ++k) {
        int off = rev ? (3 - k) : (k - 3);
        int tt = t + off;
        if (tt >= 0 && tt < N_)
            acc += w[d * DC_ + k] * xz[(size_t)(b * N_ + tt) * (2 * DI_) + d];
    }
    xi[gid] = silu_f(acc);
}

// delta = softplus(dt(16) @ dt_w(16x512) + dt_b)
__global__ void delta_kernel(const float* __restrict__ dbc,
                             const float* __restrict__ dt_w,
                             const float* __restrict__ dt_b,
                             float* __restrict__ delta) {
    int gid = blockIdx.x * 256 + threadIdx.x;   // M_*DI_
    int d = gid & (DI_ - 1);
    int r = gid >> 9;
    const float* dt = &dbc[(size_t)r * 144];
    float acc = dt_b[d];
#pragma unroll
    for (int j = 0; j < DTR_; ++j) acc += dt[j] * dt_w[j * DI_ + d];
    delta[gid] = (acc > 20.0f) ? acc : log1pf(__expf(acc));
}

// Selective scan: one (batch, channel) per lane, 64-wide state in VGPRs.
// B_t/C_t broadcast via LDS, fetched with async global->LDS copies and
// double-buffered: the next 4-step tile's 2KB DMA is issued before the
// current tile's 4x64-deep exp/FMA chains, then s_wait_asynccnt + barrier.
// rev traverses time backwards and writes at physical positions
// (== flip(mamba(flip(x))) pointwise).
__global__ __launch_bounds__(256) void scan_kernel(const float* __restrict__ delta,
                                                   const float* __restrict__ dbc,
                                                   const float* __restrict__ xi,
                                                   const float* __restrict__ A_log,
                                                   float* __restrict__ y, int rev) {
    __shared__ float sBC[2][4][128];
    const int tid = threadIdx.x;
    const int d = blockIdx.x * 256 + tid;
    const int b = blockIdx.y;
    float a[DS_], h[DS_];
#pragma unroll
    for (int s = 0; s < DS_; ++s) {
        a[s] = -__expf(A_log[(size_t)d * DS_ + s]);
        h[s] = 0.0f;
    }
    const size_t rb = (size_t)b * N_;

    // 128 lanes x b128 = 2KB per tile: dbc[(rb+p)*144 + 16 .. 143] (B then C)
    auto issue_tile = [&](int t0, int buf) {
        if (tid < 128) {
            int tt = tid >> 5;
            int j4 = (tid & 31) * 4;
            int p = rev ? (N_ - 1 - (t0 + tt)) : (t0 + tt);
            async_ld_b128(&sBC[buf][tt][j4], &dbc[(rb + p) * 144 + 16 + j4]);
        }
    };

    issue_tile(0, 0);
    wait_asynccnt0();
    __syncthreads();

    for (int t0 = 0; t0 < N_; t0 += 4) {
        const int cur = (t0 >> 2) & 1;
        if (t0 + 4 < N_) issue_tile(t0 + 4, cur ^ 1);   // overlap DMA w/ compute
#pragma unroll
        for (int tt = 0; tt < 4; ++tt) {
            int p = rev ? (N_ - 1 - (t0 + tt)) : (t0 + tt);
            size_t r = rb + p;
            float dlt = delta[r * DI_ + d];
            float xv  = xi[r * DI_ + d];
            float dx  = dlt * xv;
            float yacc = 0.0f;
            const float* Bm = sBC[cur][tt];
            const float* Cm = sBC[cur][tt] + 64;
#pragma unroll
            for (int s = 0; s < DS_; ++s) {
                float dA = __expf(dlt * a[s]);
                h[s] = h[s] * dA + dx * Bm[s];
                yacc += h[s] * Cm[s];
            }
            y[r * DI_ + d] = yacc;
        }
        wait_asynccnt0();
        __syncthreads();
    }
}

// yg = (y + xi*D) * silu(z),  z = xz[:, 512:]
__global__ void gate_kernel(const float* __restrict__ y, const float* __restrict__ xi,
                            const float* __restrict__ xz, const float* __restrict__ D,
                            float* __restrict__ yg) {
    int gid = blockIdx.x * 256 + threadIdx.x;   // M_*DI_
    int d = gid & (DI_ - 1);
    int r = gid >> 9;
    float z = xz[(size_t)r * (2 * DI_) + DI_ + d];
    yg[gid] = (y[gid] + xi[gid] * D[d]) * silu_f(z);
}

// out[r] = t1[r,:128] . cls_w2 + cls_b2   (one wave per row)
__global__ __launch_bounds__(256) void head_kernel(const float* __restrict__ t1,
                                                   const float* __restrict__ w2,
                                                   const float* __restrict__ b2,
                                                   float* __restrict__ out) {
    int lane = threadIdx.x & 31;
    int row = blockIdx.x * 8 + (threadIdx.x >> 5);
    float s = 0.0f;
#pragma unroll
    for (int i = 0; i < 4; ++i) {
        int c = i * 32 + lane;
        s += t1[(size_t)row * 128 + c] * w2[c];
    }
#pragma unroll
    for (int off = 16; off > 0; off >>= 1) s += __shfl_xor(s, off, 32);
    if (lane == 0) out[row] = s + b2[0];
}

// ================================================================= launch
extern "C" void kernel_launch(void* const* d_in, const int* in_sizes, int n_in,
                              void* d_out, int out_size, void* d_ws, size_t ws_size,
                              hipStream_t stream) {
    (void)in_sizes; (void)n_in; (void)out_size; (void)ws_size;

    const float* x    = (const float*)d_in[0];
    const int*   eidx = (const int*)d_in[1];
    // d_in[2] edge_attr, d_in[3] batch: unused by the reference.

    // params flattened in jax tree-leaf order (dict keys sorted):
    const int p = 4;
    const float* cls_b1 = (const float*)d_in[p + 0];
    const float* cls_b2 = (const float*)d_in[p + 1];
    const float* cls_w1 = (const float*)d_in[p + 2];
    const float* cls_w2 = (const float*)d_in[p + 3];
    const float* emb_b  = (const float*)d_in[p + 4];
    const float* emb_w  = (const float*)d_in[p + 5];
    const float* in_b   = (const float*)d_in[p + 6];
    const float* in_g   = (const float*)d_in[p + 7];
    const int lbase = p + 8;   // layers: [bwd(9), fwd(9), ln_b, ln_g, out_b, out_w] x 3

    // workspace layout (f32)
    float* F = (float*)d_ws;
    float* hs     = F;                               // M x 256
    float* tmp256 = hs     + (size_t)M_ * H_;        // M x 256
    float* xz     = tmp256 + (size_t)M_ * H_;        // M x 1024
    float* xi     = xz     + (size_t)M_ * 2 * DI_;   // M x 512
    float* dbc    = xi     + (size_t)M_ * DI_;       // M x 144
    float* dlt    = dbc    + (size_t)M_ * 144;       // M x 512 (delta, then yg, then t1)
    float* yb     = dlt    + (size_t)M_ * DI_;       // M x 512 (h_unsorted / y / hr)
    float* dirout = yb     + (size_t)M_ * DI_;       // M x 512 (concat fwd|bwd)
    float* deg    = dirout + (size_t)M_ * DI_;       // M
    int* sortidx  = (int*)(deg + M_);                // M
    int* undo     = sortidx + M_;                    // M

    auto gemm = [&](const float* A, int lda, const float* W, int Nn, int K,
                    const float* bias, float* C, int ldc, int colOff, int epi) {
        dim3 grid(M_ / 128, (Nn + 63) / 64);
        hipLaunchKernelGGL(gemm_bf16_wmma, grid, dim3(256), 0, stream,
                           A, lda, W, Nn, bias, C, ldc, colOff, K, Nn, epi);
    };

    // ---- degree + stable sort by degree
    hipLaunchKernelGGL(zero_kernel, dim3((M_ + 255) / 256), dim3(256), 0, stream, deg, M_);
    hipLaunchKernelGGL(degree_kernel, dim3(E_ / 256), dim3(256), 0, stream, eidx, deg);
    hipLaunchKernelGGL(sort_kernel, dim3(B_), dim3(1024), 0, stream, deg, sortidx);
    hipLaunchKernelGGL(invert_kernel, dim3((M_ + 255) / 256), dim3(256), 0, stream, sortidx, undo);

    // ---- embedding + LN + gather to sorted order
    gemm(x, IN_, emb_w, H_, IN_, emb_b, tmp256, H_, 0, 0);
    hipLaunchKernelGGL(ln_kernel, dim3(M_ / 8), dim3(256), 0, stream,
                       tmp256, (const float*)nullptr, in_g, in_b, yb);
    hipLaunchKernelGGL(gather_kernel, dim3(M_ * H_ / 256), dim3(256), 0, stream,
                       yb, sortidx, hs);

    // ---- 3 bidirectional mamba blocks
    for (int l = 0; l < 3; ++l) {
        const int base = lbase + l * 22;
        const float* ln_b_l = (const float*)d_in[base + 18];
        const float* ln_g_l = (const float*)d_in[base + 19];
        const float* out_b  = (const float*)d_in[base + 20];
        const float* out_w  = (const float*)d_in[base + 21];

        for (int dir = 0; dir < 2; ++dir) {
            // leaves per mamba (sorted): A_log, D, conv_b, conv_w, dt_b, dt_w,
            //                            in_proj, out_proj, x_proj
            const int mb = base + (dir == 0 ? 9 : 0);   // fwd first, then bwd
            const int rev = dir;                        // bwd reversed
            const int colOff = dir == 0 ? 0 : H_;       // concat [fwd | rev]
            const float* A_log    = (const float*)d_in[mb + 0];
            const float* Dp       = (const float*)d_in[mb + 1];
            const float* conv_b   = (const float*)d_in[mb + 2];
            const float* conv_w   = (const float*)d_in[mb + 3];
            const float* dt_b     = (const float*)d_in[mb + 4];
            const float* dt_w     = (const float*)d_in[mb + 5];
            const float* in_proj  = (const float*)d_in[mb + 6];
            const float* out_proj = (const float*)d_in[mb + 7];
            const float* x_proj   = (const float*)d_in[mb + 8];

            gemm(hs, H_, in_proj, 2 * DI_, H_, nullptr, xz, 2 * DI_, 0, 0);
            hipLaunchKernelGGL(conv_silu_kernel, dim3(M_ * DI_ / 256), dim3(256), 0, stream,
                               xz, conv_w, conv_b, xi, rev);
            gemm(xi, DI_, x_proj, 144, DI_, nullptr, dbc, 144, 0, 0);
            hipLaunchKernelGGL(delta_kernel, dim3(M_ * DI_ / 256), dim3(256), 0, stream,
                               dbc, dt_w, dt_b, dlt);
            hipLaunchKernelGGL(scan_kernel, dim3(DI_ / 256, B_), dim3(256), 0, stream,
                               dlt, dbc, xi, A_log, yb, rev);
            hipLaunchKernelGGL(gate_kernel, dim3(M_ * DI_ / 256), dim3(256), 0, stream,
                               yb, xi, xz, Dp, dlt);                    // yg -> dlt
            gemm(dlt, DI_, out_proj, H_, DI_, nullptr, dirout, 2 * H_, colOff, 0);
        }
        // combine + residual LayerNorm (in-place on hs is thread-local safe)
        gemm(dirout, 2 * H_, out_w, H_, 2 * H_, out_b, tmp256, H_, 0, 0);
        hipLaunchKernelGGL(ln_kernel, dim3(M_ / 8), dim3(256), 0, stream,
                           tmp256, hs, ln_g_l, ln_b_l, hs);
    }

    // ---- undo sort + classifier head
    hipLaunchKernelGGL(gather_kernel, dim3(M_ * H_ / 256), dim3(256), 0, stream,
                       hs, undo, yb);
    gemm(yb, H_, cls_w1, H_ / 2, H_, cls_b1, dlt, H_ / 2, 0, /*gelu*/1);
    hipLaunchKernelGGL(head_kernel, dim3(M_ / 8), dim3(256), 0, stream,
                       dlt, cls_w2, cls_b2, (float*)d_out);
}
